// NT_Xent_71932112274159
// MI455X (gfx1250) — matrix operand order repaired and checked by reading, not compiled, not run on hardware
//
#include <hip/hip_runtime.h>
#include <cstdint>

typedef _Float16 f16;
typedef __attribute__((ext_vector_type(8)))  _Float16 v8h;
typedef __attribute__((ext_vector_type(16))) _Float16 v16h;
typedef __attribute__((ext_vector_type(8)))  float    v8f;
typedef __attribute__((ext_vector_type(4)))  unsigned int u32x4;
typedef __attribute__((ext_vector_type(8)))  int      i32x8;
typedef __attribute__((ext_vector_type(4)))  int      i32x4;

#define DDIM 256
#define KSTEPS 8              // 256 / 32
#define TILE 16
#define WAVES 8
#define BLOCK (WAVES * 32)    // 256 threads = 8 wave32
#define ROWS_PER_WG (WAVES * TILE)  // 128 rows
#define NJCHUNK 8
#define INV_T 10.0f           // 1 / temperature
#define LDS_STRIDE 264        // 256 + 8 f16 pad -> conflict-free b128 LDS reads
#define USE_TDM 1

// ---------------------------------------------------------------- init
__global__ void ntxent_init_kernel(float* __restrict__ rowsum,
                                   float* __restrict__ out, int n) {
    int i = blockIdx.x * blockDim.x + threadIdx.x;
    if (i < n) rowsum[i] = 0.0f;
    if (i == 0) out[0] = 0.0f;
}

// ------------------------------------------------- normalize + f16 split
__global__ void ntxent_normalize_kernel(const float* __restrict__ zi,
                                        const float* __restrict__ zj,
                                        float* __restrict__ pn,
                                        f16* __restrict__ pnh,
                                        f16* __restrict__ pnl,
                                        int B, int n) {
    const int lane = threadIdx.x & 31;
    const int row  = (blockIdx.x * blockDim.x + threadIdx.x) >> 5;
    if (row >= n) return;
    const float* src = (row < B) ? (zi + (size_t)row * DDIM)
                                 : (zj + (size_t)(row - B) * DDIM);
    float x[8];
    float ss = 0.0f;
#pragma unroll
    for (int k = 0; k < 8; ++k) {
        x[k] = src[lane + 32 * k];
        ss += x[k] * x[k];
    }
    ss += __shfl_xor(ss, 1, 32);
    ss += __shfl_xor(ss, 2, 32);
    ss += __shfl_xor(ss, 4, 32);
    ss += __shfl_xor(ss, 8, 32);
    ss += __shfl_xor(ss, 16, 32);
    const float inv = 1.0f / fmaxf(sqrtf(ss), 1e-8f);
#pragma unroll
    for (int k = 0; k < 8; ++k) {
        const int c = lane + 32 * k;
        const float v = x[k] * inv;
        const f16 h = (f16)v;
        pn[(size_t)row * DDIM + c]  = v;
        pnh[(size_t)row * DDIM + c] = h;
        pnl[(size_t)row * DDIM + c] = (f16)(v - (float)h);
    }
}

// ------------------------------------------------- exact f32 positive dots
__global__ void ntxent_pos_kernel(const float* __restrict__ pn,
                                  float* __restrict__ pos, int n, int B) {
    const int lane = threadIdx.x & 31;
    const int row  = (blockIdx.x * blockDim.x + threadIdx.x) >> 5;
    if (row >= n) return;
    const int mate = (row + B) % n;
    float s = 0.0f;
#pragma unroll
    for (int k = 0; k < 8; ++k) {
        const int c = lane + 32 * k;
        s += pn[(size_t)row * DDIM + c] * pn[(size_t)mate * DDIM + c];
    }
    s += __shfl_xor(s, 1, 32);
    s += __shfl_xor(s, 2, 32);
    s += __shfl_xor(s, 4, 32);
    s += __shfl_xor(s, 8, 32);
    s += __shfl_xor(s, 16, 32);
    if (lane == 0) pos[row] = s * INV_T;
}

#if USE_TDM
// Issue one TDM 2D tile load: 16 rows x 256 f16 (512B rows, stride 512B) from
// global into LDS at lds_byte_off, with HW padding of 4 DWORDs every 128 DWORDs
// (one row) -> reproduces the 264-f16 padded LDS stride.
__device__ __forceinline__ void tdm_load_tile(const f16* gsrc, unsigned lds_byte_off) {
    const unsigned long long ga = (unsigned long long)(uintptr_t)gsrc;
    u32x4 g0;
    g0[0] = 1u;                                   // count=1, is_restore=0, gather off
    g0[1] = lds_byte_off;                         // lds_addr [63:32]
    g0[2] = (unsigned)(ga & 0xFFFFFFFFu);         // global_addr [95:64]
    g0[3] = (unsigned)((ga >> 32) & 0x01FFFFFFu)  // global_addr [120:96]
          | (2u << 30);                           // type=2 ("image") [127:126]
    i32x8 g1;
    g1[0] = (1 << 16)     // data_size = 1 -> 2 bytes/elem
          | (1 << 20)     // pad_enable
          | (6 << 22)     // pad_interval = 6 -> every 128 DWORDs (one 512B row)
          | (3 << 25);    // pad_amount = 3 -> 4 DWORDs (8 f16)
    g1[1] = (int)(256u << 16);   // tensor_dim0 = 256 (low16 at bits 63:48)
    g1[2] = (int)(16u << 16);    // tensor_dim0 hi16 = 0 | tensor_dim1 = 16 (bits 95:80)
    g1[3] = (int)(256u << 16);   // tensor_dim1 hi16 = 0 | tile_dim0 = 256 (bits 127:112)
    g1[4] = 16;                  // tile_dim1 = 16 rows | tile_dim2 = 0
    g1[5] = 256;                 // tensor_dim0_stride = 256 elems (low32)
    g1[6] = 0;                   // stride hi16 | tensor_dim1_stride low16
    g1[7] = 0;
    i32x4 g2 = {1, 1, 0, 0};     // tensor_dim2=1, tensor_dim3=1, dim2_stride=0, tile_dim3=0
    i32x4 g3 = {0, 1 << 16, 0, 0}; // dim3_stride=0, tensor_dim4=1, tile_dim4=0
#if defined(__clang_major__) && (__clang_major__ >= 23)
    i32x8 gz = {0, 0, 0, 0, 0, 0, 0, 0};
    __builtin_amdgcn_tensor_load_to_lds(g0, g1, g2, g3, gz, 0);
#else
    __builtin_amdgcn_tensor_load_to_lds(g0, g1, g2, g3, 0);
#endif
}
#endif

// ------------------------------------------------- fused WMMA sim + exp-sum
// Each wave: 16-row strip, A fragments (hi+lo, all K) resident in registers.
// Wave 0 DMAs each 16-col B tile (hi+lo) into double-buffered LDS via TDM,
// overlapping the next tile's DMA with the current tile's 24 chained WMMAs.
__global__ __launch_bounds__(BLOCK)
void ntxent_sim_kernel(const f16* __restrict__ pnh,
                       const f16* __restrict__ pnl,
                       float* __restrict__ rowsum, int n) {
    __shared__ __align__(16) f16 lds_hi[2][TILE * LDS_STRIDE];
    __shared__ __align__(16) f16 lds_lo[2][TILE * LDS_STRIDE];

    const int tid  = threadIdx.x;
    const int lane = tid & 31;
    const int wid  = tid >> 5;
    const int i0   = blockIdx.x * ROWS_PER_WG + wid * TILE;

    // ---- load A fragments (ISA 16-bit A 16x32 layout):
    // lanes 0-15: M=lane,   K chunks {0..7, 16..23}
    // lanes16-31: M=lane-16,K chunks {8..15,24..31}
    const int mrow = i0 + (lane & 15);
    const int ksel = (lane >> 4) * 8;
    v16h a_hi[KSTEPS], a_lo[KSTEPS];
#pragma unroll
    for (int kk = 0; kk < KSTEPS; ++kk) {
        const int kb = kk * 32 + ksel;
        const v8h h0 = *reinterpret_cast<const v8h*>(pnh + (size_t)mrow * DDIM + kb);
        const v8h h1 = *reinterpret_cast<const v8h*>(pnh + (size_t)mrow * DDIM + kb + 16);
        const v8h l0 = *reinterpret_cast<const v8h*>(pnl + (size_t)mrow * DDIM + kb);
        const v8h l1 = *reinterpret_cast<const v8h*>(pnl + (size_t)mrow * DDIM + kb + 16);
#pragma unroll
        for (int e = 0; e < 8; ++e) {
            a_hi[kk][e] = h0[e]; a_hi[kk][8 + e] = h1[e];
            a_lo[kk][e] = l0[e]; a_lo[kk][8 + e] = l1[e];
        }
    }

    float part[8];
#pragma unroll
    for (int v = 0; v < 8; ++v) part[v] = 0.0f;

    const int ntiles_j = n / TILE;
    const int jtper    = ntiles_j / NJCHUNK;
    const int jt0      = blockIdx.y * jtper;
    const int jtEnd    = jt0 + jtper;
    const int ncol     = lane & 15;
    const int khalf    = (lane >> 4) * 8;
    const int mofs     = (lane >> 4) << 3;   // C layout: M = v + 8*(lane>=16)

#if USE_TDM
    // prologue: DMA first tile into buffer 0
    if (wid == 0) {
        tdm_load_tile(pnh + (size_t)(jt0 * TILE) * DDIM,
                      (unsigned)(uintptr_t)&lds_hi[0][0]);
        tdm_load_tile(pnl + (size_t)(jt0 * TILE) * DDIM,
                      (unsigned)(uintptr_t)&lds_lo[0][0]);
    }
#endif

    for (int jt = jt0; jt < jtEnd; ++jt) {
        const int j0  = jt * TILE;
        const int cur = (jt - jt0) & 1;
#if USE_TDM
        if (wid == 0) {
            if (jt + 1 < jtEnd) {
                // issue next tile's DMA into the other buffer, then wait for
                // the current tile's two loads (in-order TENSORcnt)
                tdm_load_tile(pnh + (size_t)((jt + 1) * TILE) * DDIM,
                              (unsigned)(uintptr_t)&lds_hi[1 - cur][0]);
                tdm_load_tile(pnl + (size_t)((jt + 1) * TILE) * DDIM,
                              (unsigned)(uintptr_t)&lds_lo[1 - cur][0]);
                __builtin_amdgcn_s_wait_tensorcnt(2);
            } else {
                __builtin_amdgcn_s_wait_tensorcnt(0);
            }
        }
        __syncthreads();
#else
        __syncthreads();
        {
            const int row = tid >> 4;           // 0..15
            const int col = (tid & 15) * 16;    // 0..240
            *reinterpret_cast<v8h*>(&lds_hi[cur][row * LDS_STRIDE + col]) =
                *reinterpret_cast<const v8h*>(pnh + (size_t)(j0 + row) * DDIM + col);
            *reinterpret_cast<v8h*>(&lds_hi[cur][row * LDS_STRIDE + col + 8]) =
                *reinterpret_cast<const v8h*>(pnh + (size_t)(j0 + row) * DDIM + col + 8);
            *reinterpret_cast<v8h*>(&lds_lo[cur][row * LDS_STRIDE + col]) =
                *reinterpret_cast<const v8h*>(pnl + (size_t)(j0 + row) * DDIM + col);
            *reinterpret_cast<v8h*>(&lds_lo[cur][row * LDS_STRIDE + col + 8]) =
                *reinterpret_cast<const v8h*>(pnl + (size_t)(j0 + row) * DDIM + col + 8);
        }
        __syncthreads();
#endif

        v8f c = {0.f, 0.f, 0.f, 0.f, 0.f, 0.f, 0.f, 0.f};
#pragma unroll
        for (int kk = 0; kk < KSTEPS; ++kk) {
            // B 32x16 layout: lane = N, lanes 0-15 K={0..7,16..23}, lanes 16-31 K={8..15,24..31}
            const int off = ncol * LDS_STRIDE + kk * 32 + khalf;
            const v8h bh0 = *reinterpret_cast<const v8h*>(&lds_hi[cur][off]);
            const v8h bh1 = *reinterpret_cast<const v8h*>(&lds_hi[cur][off + 16]);
            const v8h bl0 = *reinterpret_cast<const v8h*>(&lds_lo[cur][off]);
            const v8h bl1 = *reinterpret_cast<const v8h*>(&lds_lo[cur][off + 16]);
            v16h b_hi, b_lo;
#pragma unroll
            for (int e = 0; e < 8; ++e) {
                b_hi[e] = bh0[e]; b_hi[8 + e] = bh1[e];
                b_lo[e] = bl0[e]; b_lo[8 + e] = bl1[e];
            }
            c = __builtin_amdgcn_wmma_f32_16x16x32_f16(false, a_hi[kk], false, b_hi,
                                                       (short)0, c, false, false);
            c = __builtin_amdgcn_wmma_f32_16x16x32_f16(false, a_hi[kk], false, b_lo,
                                                       (short)0, c, false, false);
            c = __builtin_amdgcn_wmma_f32_16x16x32_f16(false, a_lo[kk], false, b_hi,
                                                       (short)0, c, false, false);
        }

        // streaming masked exp-sum: |s| <= 10 so no max-shift needed
        const int gcol = j0 + ncol;
#pragma unroll
        for (int v = 0; v < 8; ++v) {
            const int grow = i0 + v + mofs;
            const float e = __expf(c[v] * INV_T);
            part[v] += (grow != gcol) ? e : 0.0f;
        }
        __syncthreads();   // protect buf[cur] before it is re-DMAed next iter
    }

    // reduce over the 16 lanes of each half (N dim), then combine across blocks
#pragma unroll
    for (int v = 0; v < 8; ++v) {
        float s = part[v];
        s += __shfl_xor(s, 1, 32);
        s += __shfl_xor(s, 2, 32);
        s += __shfl_xor(s, 4, 32);
        s += __shfl_xor(s, 8, 32);
        if ((lane & 15) == 0) {
            const int grow = i0 + v + mofs;
            atomicAdd(&rowsum[grow], s);
        }
    }
}

// ------------------------------------------------- loss = mean(log(sum) - pos)
__global__ void ntxent_finalize_kernel(const float* __restrict__ rowsum,
                                       const float* __restrict__ pos,
                                       float* __restrict__ out, int n) {
    const int i = blockIdx.x * blockDim.x + threadIdx.x;
    float v = 0.0f;
    if (i < n) v = (logf(rowsum[i]) - pos[i]) / (float)n;
    v += __shfl_xor(v, 1, 32);
    v += __shfl_xor(v, 2, 32);
    v += __shfl_xor(v, 4, 32);
    v += __shfl_xor(v, 8, 32);
    v += __shfl_xor(v, 16, 32);
    if ((threadIdx.x & 31) == 0) atomicAdd(out, v);
}

extern "C" void kernel_launch(void* const* d_in, const int* in_sizes, int n_in,
                              void* d_out, int out_size, void* d_ws, size_t ws_size,
                              hipStream_t stream) {
    const float* zi = (const float*)d_in[0];
    const float* zj = (const float*)d_in[1];
    float* out = (float*)d_out;
    const int B = in_sizes[0] / DDIM;   // 4096
    const int n = 2 * B;                // 8192

    // workspace layout (~16.1 MB): pn f32 | pn_hi f16 | pn_lo f16 | pos | rowsum
    char* w = (char*)d_ws;
    float* pn  = (float*)w;  w += (size_t)n * DDIM * sizeof(float);
    f16* pnh   = (f16*)w;    w += (size_t)n * DDIM * sizeof(f16);
    f16* pnl   = (f16*)w;    w += (size_t)n * DDIM * sizeof(f16);
    float* pos = (float*)w;  w += (size_t)n * sizeof(float);
    float* rowsum = (float*)w;

    ntxent_init_kernel<<<(n + 255) / 256, 256, 0, stream>>>(rowsum, out, n);
    ntxent_normalize_kernel<<<(n + 7) / 8, 256, 0, stream>>>(zi, zj, pn, pnh, pnl, B, n);
    ntxent_pos_kernel<<<(n + 7) / 8, 256, 0, stream>>>(pn, pos, n, B);
    dim3 grid(n / ROWS_PER_WG, NJCHUNK);
    ntxent_sim_kernel<<<grid, BLOCK, 0, stream>>>(pnh, pnl, rowsum, n);
    ntxent_finalize_kernel<<<(n + 255) / 256, 256, 0, stream>>>(rowsum, pos, out, n);
}
